// Llama4DecoderLayer_17506286698806
// MI455X (gfx1250) — matrix operand order repaired and checked by reading, not compiled, not run on hardware
//
#include <hip/hip_runtime.h>

#define T_TOK 2048
#define H_DIM 1024
#define NH 16
#define NKV 4
#define HD 64
#define NE 8
#define FF 1024
#define QKV_DIM ((NH + 2 * NKV) * HD) /* 1536 */
#define EPSV 1e-5f
#define THETA_V 500000.0f
#define SCALEV 0.125f
#define PAD_ROWS 3072
#define PAD_TILES (PAD_ROWS / 128)

#if defined(__gfx1250__) && __has_builtin(__builtin_amdgcn_global_load_async_to_lds_b128)
#define ASYNC_LDS_OK 1
#else
#define ASYNC_LDS_OK 0
#endif

typedef __attribute__((ext_vector_type(16))) __bf16 v16bf;
typedef __attribute__((ext_vector_type(8)))  __bf16 v8bf;
typedef __attribute__((ext_vector_type(8)))  float  v8f;

__device__ __forceinline__ v16bf mk16(v8bf lo, v8bf hi) {
  v16bf r;
#pragma unroll
  for (int i = 0; i < 8; ++i) { r[i] = lo[i]; r[i + 8] = hi[i]; }
  return r;
}
__device__ __forceinline__ float wred_sum(float v) {
#pragma unroll
  for (int o = 16; o >= 1; o >>= 1) v += __shfl_xor(v, o, 32);
  return v;
}
// reductions within a 16-lane half of the wave (rows of a WMMA C fragment)
__device__ __forceinline__ float hred_max(float v) {
#pragma unroll
  for (int o = 8; o >= 1; o >>= 1) v = fmaxf(v, __shfl_xor(v, o, 32));
  return v;
}
__device__ __forceinline__ float hred_sum(float v) {
#pragma unroll
  for (int o = 8; o >= 1; o >>= 1) v += __shfl_xor(v, o, 32);
  return v;
}

#if ASYNC_LDS_OK
// Builtin signature (from clang diagnostic): arg0 = GCC-vector int4 pointer in
// the device/global address space (AS1), arg1 = LDS (AS3), then imm offset and
// imm cpol. Build the AS pointers via integer casts: global addresses are
// identity-mapped in the flat aperture; generic LDS pointers carry the LDS
// offset in their low 32 bits.
typedef int v4i_vs __attribute__((vector_size(16)));
typedef __attribute__((address_space(1))) v4i_vs* glb128_t;
typedef __attribute__((address_space(3))) v4i_vs* lds128_t;

__device__ __forceinline__ void async_ld_b128(const void* gsrc, void* ldst) {
  __builtin_amdgcn_global_load_async_to_lds_b128(
      (glb128_t)(unsigned long long)gsrc,
      (lds128_t)(unsigned int)(unsigned long long)ldst, 0, 0);
}
#endif

// ---------------- fp32 -> bf16 conversion (one-time weight pass) -------------
__global__ void cvt_bf16_kernel(const float* __restrict__ in, __bf16* __restrict__ out,
                                long long n) {
  long long i  = (long long)blockIdx.x * blockDim.x + threadIdx.x;
  long long st = (long long)gridDim.x * blockDim.x;
  for (; i < n; i += st) out[i] = (__bf16)in[i];
}

// ---------------- rmsnorm (fp32 in, bf16 out) --------------------------------
__global__ __launch_bounds__(256) void rmsnorm_kernel(const float* __restrict__ x,
                                                      const float* __restrict__ w,
                                                      __bf16* __restrict__ out) {
  const int t = blockIdx.x;
  const float* xr = x + (long long)t * H_DIM;
  float ss = 0.f;
  for (int j = threadIdx.x; j < H_DIM; j += 256) { float vv = xr[j]; ss += vv * vv; }
  ss = wred_sum(ss);
  __shared__ float red[8];
  if ((threadIdx.x & 31) == 0) red[threadIdx.x >> 5] = ss;
  __syncthreads();
  float tot = 0.f;
#pragma unroll
  for (int i = 0; i < 8; ++i) tot += red[i];
  const float sc = rsqrtf(tot * (1.0f / H_DIM) + EPSV);
  for (int j = threadIdx.x; j < H_DIM; j += 256)
    out[(long long)t * H_DIM + j] = (__bf16)(xr[j] * sc * w[j]);
}

// ---------------- unified bf16 WMMA GEMM -------------------------------------
// C[M x N] = gatherA(A)[M x K] * B[K x N] (+ Cadd), rows optionally scattered
// via gatherC; per-M-tile expert B selection via tileExpert/bExpertStride.
// BM=128, BN=64, BK=64, 256 threads (8 waves, 4x2 of 32x32 per-wave tiles).
// A tile staged via GLOBAL_LOAD_ASYNC_TO_LDS_B128 when available.
__global__ __launch_bounds__(256) void gemm_bf16_kernel(
    const __bf16* __restrict__ A, const __bf16* __restrict__ Bw,
    float* __restrict__ C, const float* __restrict__ Cadd,
    const int* __restrict__ gatherA, const int* __restrict__ gatherC,
    const int* __restrict__ tileExpert, long long bExpertStride,
    int N, int K) {
  __shared__ __bf16 Alds[128][64];
  __shared__ __bf16 Blds[64][64]; // stored transposed: [n][k]
  const int mt = blockIdx.y, nt = blockIdx.x;
  const __bf16* Bp = Bw;
  if (tileExpert) {
    int e = tileExpert[mt];
    if (e < 0) return;
    Bp += (long long)e * bExpertStride;
  }
  const int tid  = threadIdx.x;
  const int wave = tid >> 5, lane = tid & 31;
  const int wm = (wave >> 1) * 32, wn = (wave & 1) * 32;
  const int lr  = lane & 15;
  const int alk = (lane < 16) ? 0 : 8;   // A-frag K sub-chunk
  const int bk  = (lane < 16) ? 0 : 8;   // B-frag K sub-chunk
  v8f acc[2][2] = {};
  const int m0 = mt * 128;
  for (int k0 = 0; k0 < K; k0 += 64) {
#pragma unroll
    for (int i = 0; i < 4; ++i) { // A tile: 128x64, one 16B chunk x4 per thread
      int c = tid + i * 256;
      int r = c >> 3, kc = (c & 7) * 8;
      int arow = m0 + r;
      if (gatherA) { int g = gatherA[arow]; arow = (g < 0) ? 0 : g; }
      const __bf16* src = A + (long long)arow * K + k0 + kc;
#if ASYNC_LDS_OK
      async_ld_b128(src, &Alds[r][kc]);
#else
      *(v8bf*)&Alds[r][kc] = *(const v8bf*)src;
#endif
    }
#pragma unroll
    for (int i = 0; i < 2; ++i) { // B tile: 64x64 global -> [n][k] LDS
      int c = tid + i * 256;
      int r = c >> 3, nc = (c & 7) * 8;
      const __bf16* bsrc = Bp + (long long)(k0 + r) * N + (long long)nt * 64 + nc;
      v8bf v = *(const v8bf*)bsrc;
      if (k0 + 64 < K) __builtin_prefetch(bsrc + (long long)64 * N, 0, 1);
#pragma unroll
      for (int j = 0; j < 8; ++j) Blds[nc + j][r] = v[j];
    }
#if ASYNC_LDS_OK
    asm volatile("s_wait_asynccnt 0" ::: "memory");
#endif
    __syncthreads();
#pragma unroll
    for (int ks = 0; ks < 64; ks += 32) {
      v16bf af[2], bfr[2];
#pragma unroll
      for (int i = 0; i < 2; ++i)
        af[i] = mk16(*(const v8bf*)&Alds[wm + i * 16 + lr][ks + alk],
                     *(const v8bf*)&Alds[wm + i * 16 + lr][ks + 16 + alk]);
#pragma unroll
      for (int j = 0; j < 2; ++j)
        bfr[j] = mk16(*(const v8bf*)&Blds[wn + j * 16 + lr][ks + bk],
                      *(const v8bf*)&Blds[wn + j * 16 + lr][ks + bk + 16]);
#pragma unroll
      for (int i = 0; i < 2; ++i)
#pragma unroll
        for (int j = 0; j < 2; ++j)
          acc[i][j] = __builtin_amdgcn_wmma_f32_16x16x32_bf16(
              false, af[i], false, bfr[j], (short)0, acc[i][j], false, false);
    }
    __syncthreads();
  }
  const int lh = lane >> 4;
#pragma unroll
  for (int i = 0; i < 2; ++i)
#pragma unroll
    for (int j = 0; j < 2; ++j)
#pragma unroll
      for (int r = 0; r < 8; ++r) {
        int row  = m0 + wm + i * 16 + lh * 8 + r;
        int crow = row;
        if (gatherC) { int g = gatherC[row]; if (g < 0) continue; crow = g; }
        int col = nt * 64 + wn + j * 16 + lr;
        float v = acc[i][j][r];
        if (Cadd) v += Cadd[(long long)crow * N + col];
        C[(long long)crow * N + col] = v;
      }
}

// ---------------- RoPE + q/k rmsnorm + v convert -----------------------------
__global__ __launch_bounds__(32) void rope_kernel(
    const float* __restrict__ qkv, const int* __restrict__ pos,
    __bf16* __restrict__ qo, __bf16* __restrict__ ko, __bf16* __restrict__ vo) {
  const int t = blockIdx.x, hy = blockIdx.y, lane = threadIdx.x;
  const float x1 = qkv[(long long)t * QKV_DIM + hy * HD + lane];
  const float x2 = qkv[(long long)t * QKV_DIM + hy * HD + lane + 32];
  if (hy >= NH + NKV) { // value head: plain bf16 convert
    __bf16* dst = vo + ((long long)t * NKV + (hy - NH - NKV)) * HD;
    dst[lane] = (__bf16)x1; dst[lane + 32] = (__bf16)x2;
    return;
  }
  const float p   = (float)pos[t];
  const float inv = __expf(-((float)lane * (1.0f / 32.0f)) * __logf(THETA_V));
  float sn, cs;
  __sincosf(p * inv, &sn, &cs);
  float y1 = x1 * cs - x2 * sn;
  float y2 = x2 * cs + x1 * sn;
  float ss = wred_sum(y1 * y1 + y2 * y2);
  const float sc = rsqrtf(ss * (1.0f / 64.0f) + EPSV);
  y1 *= sc; y2 *= sc;
  __bf16* dst = (hy < NH) ? (qo + ((long long)t * NH + hy) * HD)
                          : (ko + ((long long)t * NKV + (hy - NH)) * HD);
  dst[lane] = (__bf16)y1; dst[lane + 32] = (__bf16)y2;
}

// ---------------- flash attention (causal GQA), WMMA for QK^T and PV ---------
__global__ __launch_bounds__(256) void attn_kernel(
    const __bf16* __restrict__ q, const __bf16* __restrict__ k,
    const __bf16* __restrict__ v, __bf16* __restrict__ ao) {
  __shared__ __bf16 Pls[8][16][32];  // per-wave P repack (C-layout -> A-layout)
  __shared__ __bf16 Vt[8][64][40];   // per-wave V^T [d][key], padded stride
  const int w = threadIdx.x >> 5, lane = threadIdx.x & 31;
  const int qb = blockIdx.x * 8 + w;
  const int h  = blockIdx.y;
  const int kh = h >> 2; // GQA: 4 q-heads per kv-head
  const int q0 = qb * 16;
  const int lr = lane & 15, lh = lane >> 4;
  const int alk = lh ? 8 : 0;
  const int bk  = lh ? 8 : 0;

  v16bf aq[2];
  {
    long long qbase = ((long long)(q0 + lr) * NH + h) * HD;
#pragma unroll
    for (int dsi = 0; dsi < 2; ++dsi)
      aq[dsi] = mk16(*(const v8bf*)(q + qbase + dsi * 32 + alk),
                     *(const v8bf*)(q + qbase + dsi * 32 + 16 + alk));
  }
  v8f o[4] = {};
  float mrow[8], lrow[8];
#pragma unroll
  for (int r = 0; r < 8; ++r) { mrow[r] = -1e30f; lrow[r] = 0.f; }

  const int kbmax = (q0 + 15) >> 5;
  for (int kb = 0; kb <= kbmax; ++kb) {
    const int k0 = kb * 32;
    v8f s[2] = {};
#pragma unroll
    for (int f = 0; f < 2; ++f) {
      long long kbase = ((long long)(k0 + f * 16 + lr) * NKV + kh) * HD;
#pragma unroll
      for (int dsi = 0; dsi < 2; ++dsi) {
        int dk = dsi * 32 + bk;
        v16bf bkf = mk16(*(const v8bf*)(k + kbase + dk),
                         *(const v8bf*)(k + kbase + dk + 16));
        s[f] = __builtin_amdgcn_wmma_f32_16x16x32_bf16(
            false, aq[dsi], false, bkf, (short)0, s[f], false, false);
      }
    }
    // online softmax update
#pragma unroll
    for (int r = 0; r < 8; ++r) {
      const int row = q0 + lh * 8 + r;
      float s0 = s[0][r] * SCALEV;
      float s1 = s[1][r] * SCALEV;
      if (k0 + lr > row)      s0 = -1e30f;
      if (k0 + 16 + lr > row) s1 = -1e30f;
      float mx = hred_max(fmaxf(s0, s1));
      float mn = fmaxf(mrow[r], mx);
      float p0 = __expf(s0 - mn), p1 = __expf(s1 - mn);
      float sum   = hred_sum(p0 + p1);
      float alpha = __expf(mrow[r] - mn);
      lrow[r] = lrow[r] * alpha + sum;
      mrow[r] = mn;
#pragma unroll
      for (int ntl = 0; ntl < 4; ++ntl) o[ntl][r] *= alpha;
      Pls[w][lh * 8 + r][lr]      = (__bf16)p0;
      Pls[w][lh * 8 + r][16 + lr] = (__bf16)p1;
    }
    { // stage V (32 keys x 64 d) transposed into LDS
      long long vbase = ((long long)(k0 + lane) * NKV + kh) * HD;
#pragma unroll
      for (int c = 0; c < 64; c += 8) {
        v8bf vv = *(const v8bf*)(v + vbase + c);
#pragma unroll
        for (int j = 0; j < 8; ++j) Vt[w][c + j][lane] = vv[j];
      }
    }
    asm volatile("s_wait_dscnt 0" ::: "memory"); // wave-local LDS RAW
    v16bf ap = mk16(*(const v8bf*)&Pls[w][lr][alk],
                    *(const v8bf*)&Pls[w][lr][16 + alk]);
#pragma unroll
    for (int ntl = 0; ntl < 4; ++ntl) {
      v16bf bv = mk16(*(const v8bf*)&Vt[w][ntl * 16 + lr][bk],
                      *(const v8bf*)&Vt[w][ntl * 16 + lr][bk + 16]);
      o[ntl] = __builtin_amdgcn_wmma_f32_16x16x32_bf16(
          false, ap, false, bv, (short)0, o[ntl], false, false);
    }
  }
#pragma unroll
  for (int ntl = 0; ntl < 4; ++ntl)
#pragma unroll
    for (int r = 0; r < 8; ++r) {
      int row = q0 + lh * 8 + r;
      float val = o[ntl][r] / lrow[r];
      ao[((long long)row * NH + h) * HD + ntl * 16 + lr] = (__bf16)val;
    }
}

// ---------------- router: logits, top-1, sigmoid gate, x_in = h*gate ---------
__global__ __launch_bounds__(256) void router_kernel(
    const __bf16* __restrict__ h2, const float* __restrict__ rw,
    int* __restrict__ idx, __bf16* __restrict__ xin) {
  const int t = blockIdx.x;
  float p[NE];
#pragma unroll
  for (int e = 0; e < NE; ++e) p[e] = 0.f;
  for (int j = threadIdx.x; j < H_DIM; j += 256) {
    float hv = (float)h2[(long long)t * H_DIM + j];
#pragma unroll
    for (int e = 0; e < NE; ++e) p[e] += hv * rw[(long long)j * NE + e];
  }
  __shared__ float red[NE][8];
  __shared__ float gsh;
  const int wv = threadIdx.x >> 5, lane = threadIdx.x & 31;
#pragma unroll
  for (int e = 0; e < NE; ++e) {
    float s = wred_sum(p[e]);
    if (lane == 0) red[e][wv] = s;
  }
  __syncthreads();
  if (threadIdx.x == 0) {
    float best = -1e30f; int bi = 0;
#pragma unroll
    for (int e = 0; e < NE; ++e) {
      float s = 0.f;
#pragma unroll
      for (int i2 = 0; i2 < 8; ++i2) s += red[e][i2];
      if (s > best) { best = s; bi = e; }
    }
    idx[t] = bi;
    gsh = 1.0f / (1.0f + __expf(-best));
  }
  __syncthreads();
  const float g = gsh;
  for (int j = threadIdx.x; j < H_DIM; j += 256)
    xin[(long long)t * H_DIM + j] =
        (__bf16)((float)h2[(long long)t * H_DIM + j] * g);
}

// ---------------- MoE token permutation (expert-sorted, 128-aligned segs) ----
__global__ void moe_zero_kernel(int* c) { if (threadIdx.x < 16) c[threadIdx.x] = 0; }

__global__ void moe_count_kernel(const int* __restrict__ idx, int* __restrict__ counts) {
  int t = blockIdx.x * blockDim.x + threadIdx.x;
  if (t < T_TOK) atomicAdd(&counts[idx[t]], 1);
}

__global__ __launch_bounds__(256) void moe_offsets_kernel(
    const int* __restrict__ counts, int* __restrict__ base,
    int* __restrict__ tileExpert, int* __restrict__ perm, int* __restrict__ cursors) {
  for (int i = threadIdx.x; i < PAD_ROWS; i += 256) perm[i] = -1;
  if (threadIdx.x == 0) {
    int b = 0;
    for (int e = 0; e < NE; ++e) {
      base[e] = b;
      int padded = ((counts[e] + 127) >> 7) << 7;
      for (int tt = b >> 7; tt < (b + padded) >> 7; ++tt) tileExpert[tt] = e;
      b += padded;
      cursors[e] = 0;
    }
    for (int tt = b >> 7; tt < PAD_TILES; ++tt) tileExpert[tt] = -1;
  }
}

__global__ void moe_scatter_kernel(const int* __restrict__ idx, const int* __restrict__ base,
                                   int* __restrict__ cursors, int* __restrict__ perm) {
  int t = blockIdx.x * blockDim.x + threadIdx.x;
  if (t < T_TOK) {
    int e = idx[t];
    int r = atomicAdd(&cursors[e], 1);
    perm[base[e] + r] = t;
  }
}

// ---------------- silu(a) * b -> bf16 ----------------------------------------
__global__ void silu_mul_kernel(const float* __restrict__ a, const float* __restrict__ b,
                                __bf16* __restrict__ g, long long n) {
  long long i  = (long long)blockIdx.x * blockDim.x + threadIdx.x;
  long long st = (long long)gridDim.x * blockDim.x;
  for (; i < n; i += st) {
    float x = a[i];
    g[i] = (__bf16)((x / (1.0f + __expf(-x))) * b[i]);
  }
}

// =============================================================================
extern "C" void kernel_launch(void* const* d_in, const int* in_sizes, int n_in,
                              void* d_out, int out_size, void* d_ws, size_t ws_size,
                              hipStream_t stream) {
  (void)in_sizes; (void)n_in; (void)out_size; (void)ws_size;
  const int*   positions = (const int*)  d_in[0];
  const float* hidden    = (const float*)d_in[1];
  const float* ln1_w     = (const float*)d_in[2];
  const float* ln2_w     = (const float*)d_in[3];
  const float* wqkv      = (const float*)d_in[4];
  const float* wo        = (const float*)d_in[5];
  const float* router_w  = (const float*)d_in[6];
  const float* w_gate    = (const float*)d_in[7];
  const float* w_up      = (const float*)d_in[8];
  const float* w_down    = (const float*)d_in[9];
  const float* sh_gate   = (const float*)d_in[10];
  const float* sh_up     = (const float*)d_in[11];
  const float* sh_down   = (const float*)d_in[12];

  float* outF   = (float*)d_out;                       // [T, H] MoE+shared output
  float* residF = outF + (size_t)T_TOK * H_DIM;        // [T, H] residual output

  char* ws = (char*)d_ws;
  size_t off = 0;
  auto alloc = [&](size_t bytes) -> void* {
    void* p = ws + off;
    off += (bytes + 255) & ~(size_t)255;
    return p;
  };
  __bf16* wqkv_b = (__bf16*)alloc((size_t)H_DIM * QKV_DIM * 2);
  __bf16* wo_b   = (__bf16*)alloc((size_t)NH * HD * H_DIM * 2);
  __bf16* wg_b   = (__bf16*)alloc((size_t)NE * H_DIM * FF * 2);
  __bf16* wu_b   = (__bf16*)alloc((size_t)NE * H_DIM * FF * 2);
  __bf16* wd_b   = (__bf16*)alloc((size_t)NE * FF * H_DIM * 2);
  __bf16* shg_b  = (__bf16*)alloc((size_t)H_DIM * FF * 2);
  __bf16* shu_b  = (__bf16*)alloc((size_t)H_DIM * FF * 2);
  __bf16* shd_b  = (__bf16*)alloc((size_t)FF * H_DIM * 2);
  __bf16* h1_b   = (__bf16*)alloc((size_t)T_TOK * H_DIM * 2);
  float*  qkv_f  = (float*) alloc((size_t)T_TOK * QKV_DIM * 4);
  __bf16* q_b    = (__bf16*)alloc((size_t)T_TOK * NH * HD * 2);
  __bf16* k_b    = (__bf16*)alloc((size_t)T_TOK * NKV * HD * 2);
  __bf16* v_b    = (__bf16*)alloc((size_t)T_TOK * NKV * HD * 2);
  __bf16* ao_b   = (__bf16*)alloc((size_t)T_TOK * NH * HD * 2);
  __bf16* h2_b   = (__bf16*)alloc((size_t)T_TOK * H_DIM * 2);
  __bf16* xin_b  = (__bf16*)alloc((size_t)T_TOK * H_DIM * 2);
  __bf16* g_b    = (__bf16*)alloc((size_t)PAD_ROWS * FF * 2);
  float*  t1     = (float*) alloc((size_t)PAD_ROWS * FF * 4);
  float*  t2     = (float*) alloc((size_t)PAD_ROWS * FF * 4);
  int*    idx    = (int*)   alloc((size_t)T_TOK * 4);
  int*    counts = (int*)   alloc(16 * 4);
  int*    cursors= (int*)   alloc(16 * 4);
  int*    base   = (int*)   alloc(16 * 4);
  int*    perm   = (int*)   alloc((size_t)PAD_ROWS * 4);
  int*    tileE  = (int*)   alloc((size_t)PAD_TILES * 4);

  auto cvt = [&](const float* src, __bf16* dst, long long n) {
    int blocks = (int)((n + 255) / 256);
    if (blocks > 4096) blocks = 4096;
    cvt_bf16_kernel<<<blocks, 256, 0, stream>>>(src, dst, n);
  };
  auto gemm = [&](const __bf16* A, const __bf16* B, float* C, const float* Cadd,
                  const int* gA, const int* gC, const int* tE, long long bStride,
                  int M, int N, int K) {
    gemm_bf16_kernel<<<dim3(N / 64, M / 128), 256, 0, stream>>>(
        A, B, C, Cadd, gA, gC, tE, bStride, N, K);
  };

  // 1. one-time weight fp32->bf16 conversion (weights then live in L2)
  cvt(wqkv,    wqkv_b, (long long)H_DIM * QKV_DIM);
  cvt(wo,      wo_b,   (long long)NH * HD * H_DIM);
  cvt(w_gate,  wg_b,   (long long)NE * H_DIM * FF);
  cvt(w_up,    wu_b,   (long long)NE * H_DIM * FF);
  cvt(w_down,  wd_b,   (long long)NE * FF * H_DIM);
  cvt(sh_gate, shg_b,  (long long)H_DIM * FF);
  cvt(sh_up,   shu_b,  (long long)H_DIM * FF);
  cvt(sh_down, shd_b,  (long long)FF * H_DIM);

  // 2. attention block
  rmsnorm_kernel<<<T_TOK, 256, 0, stream>>>(hidden, ln1_w, h1_b);
  gemm(h1_b, wqkv_b, qkv_f, nullptr, nullptr, nullptr, nullptr, 0,
       T_TOK, QKV_DIM, H_DIM);
  rope_kernel<<<dim3(T_TOK, NH + 2 * NKV), 32, 0, stream>>>(qkv_f, positions,
                                                            q_b, k_b, v_b);
  attn_kernel<<<dim3(T_TOK / 128, NH), 256, 0, stream>>>(q_b, k_b, v_b, ao_b);
  gemm(ao_b, wo_b, residF, hidden, nullptr, nullptr, nullptr, 0,
       T_TOK, H_DIM, NH * HD); // residF = ao@wo + hidden  (residual output)

  // 3. MLP block
  rmsnorm_kernel<<<T_TOK, 256, 0, stream>>>(residF, ln2_w, h2_b);

  // shared expert -> outF
  gemm(h2_b, shg_b, t1, nullptr, nullptr, nullptr, nullptr, 0, T_TOK, FF, H_DIM);
  gemm(h2_b, shu_b, t2, nullptr, nullptr, nullptr, nullptr, 0, T_TOK, FF, H_DIM);
  silu_mul_kernel<<<4096, 256, 0, stream>>>(t1, t2, g_b, (long long)T_TOK * FF);
  gemm(g_b, shd_b, outF, nullptr, nullptr, nullptr, nullptr, 0, T_TOK, H_DIM, FF);

  // routed expert (top-1): sort tokens by expert into 128-aligned segments
  router_kernel<<<T_TOK, 256, 0, stream>>>(h2_b, router_w, idx, xin_b);
  moe_zero_kernel<<<1, 32, 0, stream>>>(counts);
  moe_count_kernel<<<T_TOK / 256, 256, 0, stream>>>(idx, counts);
  moe_offsets_kernel<<<1, 256, 0, stream>>>(counts, base, tileE, perm, cursors);
  moe_scatter_kernel<<<T_TOK / 256, 256, 0, stream>>>(idx, base, cursors, perm);

  gemm(xin_b, wg_b, t1, nullptr, perm, nullptr, tileE,
       (long long)H_DIM * FF, PAD_ROWS, FF, H_DIM);
  gemm(xin_b, wu_b, t2, nullptr, perm, nullptr, tileE,
       (long long)H_DIM * FF, PAD_ROWS, FF, H_DIM);
  silu_mul_kernel<<<4096, 256, 0, stream>>>(t1, t2, g_b, (long long)PAD_ROWS * FF);
  gemm(g_b, wd_b, outF, outF, nullptr, perm, tileE,
       (long long)FF * H_DIM, PAD_ROWS, H_DIM, FF); // outF += routed (scatter)
}
